// ScanAttention_14130442404169
// MI455X (gfx1250) — matrix operand order repaired and checked by reading, not compiled, not run on hardware
//
#include <hip/hip_runtime.h>

typedef __attribute__((ext_vector_type(16))) _Float16 v16h;
typedef __attribute__((ext_vector_type(8)))  _Float16 v8h;
typedef __attribute__((ext_vector_type(2)))  _Float16 v2h;
typedef __attribute__((ext_vector_type(8)))  float    v8f;

#define B_   2
#define Q_   2048
#define K_   2048
#define H_   16
#define D_   64
#define QT   128          // query rows per workgroup
#define KB   64           // keys per KV block
#define NW   8            // waves per workgroup (256 threads, wave32)
// scale = 1/sqrt(64) * log2(e): scores land directly in log2 domain -> exp2f
#define QSCALE 0.18033688011112042f
#define NEG_BIG (-3.0e38f)

// Packed 16-lane max butterfly: 4 DWORDs carry 8 rows as f16x2
// (v_pk_max_num_f16). All lanes in each 16-lane half end with identical
// packed results.
__device__ __forceinline__ void hmax16x2(int (&pk)[4]) {
  #pragma unroll
  for (int off = 1; off <= 8; off <<= 1) {
    #pragma unroll
    for (int i = 0; i < 4; ++i) {
      const int o = __shfl_xor(pk[i], off, 32);
      const v2h a = __builtin_bit_cast(v2h, pk[i]);
      const v2h b = __builtin_bit_cast(v2h, o);
      pk[i] = __builtin_bit_cast(int, __builtin_elementwise_max(a, b));
    }
  }
}

// Stage one 64-key block: K transposed [d][key] (pair-packed b32 stores),
// V row-major [key][d] (b128 stores). 256 threads cooperate.
__device__ __forceinline__ void stage_kv(const float* __restrict__ ks,
                                         const float* __restrict__ vs,
                                         _Float16* __restrict__ kt,
                                         _Float16* __restrict__ v,
                                         int tid, long krow0, int h)
{
  // K: thread handles keys {2*kp, 2*kp+1} x 8 d-values -> 8 packed b32 stores
  {
    const int kp   = tid & 31;        // key pair index
    const int dseg = (tid >> 5) * 8;  // 8 d-values
    const int key0 = 2 * kp;
    const float4* r0 = (const float4*)(ks + ((krow0 + key0    )*H_ + h)*D_ + dseg);
    const float4* r1 = (const float4*)(ks + ((krow0 + key0 + 1)*H_ + h)*D_ + dseg);
    float4 a0 = r0[0], a1 = r0[1];
    float4 b0 = r1[0], b1 = r1[1];
    float fa[8] = {a0.x,a0.y,a0.z,a0.w,a1.x,a1.y,a1.z,a1.w};
    float fb[8] = {b0.x,b0.y,b0.z,b0.w,b1.x,b1.y,b1.z,b1.w};
    #pragma unroll
    for (int i = 0; i < 8; ++i) {
      v2h pk;
      pk[0] = (_Float16)fa[i];
      pk[1] = (_Float16)fb[i];
      *(v2h*)&kt[(dseg + i)*KB + key0] = pk;
    }
  }
  // V: thread handles 1 key x 16 d-values -> 2 b128 stores
  {
    const int key = tid >> 2;
    const int seg = (tid & 3) * 16;
    const float4* vsrc = (const float4*)(vs + ((krow0 + key)*H_ + h)*D_ + seg);
    v8h* vdst = (v8h*)&v[key * D_ + seg];
    #pragma unroll
    for (int i = 0; i < 2; ++i) {
      float4 f0 = vsrc[2*i+0], f1 = vsrc[2*i+1];
      v8h hv;
      hv[0]=(_Float16)f0.x; hv[1]=(_Float16)f0.y; hv[2]=(_Float16)f0.z; hv[3]=(_Float16)f0.w;
      hv[4]=(_Float16)f1.x; hv[5]=(_Float16)f1.y; hv[6]=(_Float16)f1.z; hv[7]=(_Float16)f1.w;
      vdst[i] = hv;
    }
  }
}

// One KV block of flash attention for this wave's 16 query rows.
// MASKED=false is the hot path (block fully inside valid_len): no cndmask work.
// Denominator accumulates on the WMMA pipe (P x ones); row max uses a packed
// f16x2 butterfly (online softmax is exactly invariant to the nm used, so f16
// precision for the max is safe).
template <bool MASKED>
__device__ __forceinline__ void kv_block(const _Float16* __restrict__ sKT,
                                         const _Float16* __restrict__ sV,
                                         _Float16* __restrict__ pw,
                                         const v16h (&aQ)[2], const v16h& bones,
                                         v8f (&Oacc)[4], v8f& Lacc,
                                         float (&m)[8],
                                         int lane, int hi8, int kb, int valid)
{
  // ---- S = (Q*scale*log2e) K^T ----
  v8f S[4];
  #pragma unroll
  for (int t = 0; t < 4; ++t) {
    v8f st = {};
    #pragma unroll
    for (int c = 0; c < 2; ++c) {
      // B-frag: lane holds key = t*16 + lane%16; elements d in table order
      v16h bk;
      #pragma unroll
      for (int s = 0; s < 16; ++s)
        bk[s] = sKT[(c*32 + hi8 + (s&7) + ((s>>3)<<4))*KB + t*16 + (lane & 15)];
      st = __builtin_amdgcn_wmma_f32_16x16x32_f16(false, aQ[c], false, bk,
                                                  (short)0, st, false, false);
    }
    S[t] = st;
  }

  if (MASKED) {
    #pragma unroll
    for (int t = 0; t < 4; ++t) {
      const bool ok = (kb + t*16 + (lane & 15)) < valid;
      #pragma unroll
      for (int j = 0; j < 8; ++j) S[t][j] = ok ? S[t][j] : NEG_BIG;
    }
  }

  // ---- per-row local max, then packed cross-lane butterfly ----
  float lm[8];
  #pragma unroll
  for (int j = 0; j < 8; ++j)
    lm[j] = fmaxf(fmaxf(S[0][j], S[1][j]), fmaxf(S[2][j], S[3][j]));

  int pk[4];
  #pragma unroll
  for (int i = 0; i < 4; ++i) {
    v2h hp;
    hp[0] = (_Float16)lm[2*i];
    hp[1] = (_Float16)lm[2*i+1];
    pk[i] = __builtin_bit_cast(int, hp);
  }
  hmax16x2(pk);

  float mx[8];
  #pragma unroll
  for (int i = 0; i < 4; ++i) {
    const v2h hp = __builtin_bit_cast(v2h, pk[i]);
    mx[2*i]   = (float)hp[0];
    mx[2*i+1] = (float)hp[1];
  }

  // ---- online softmax in log2 domain ----
  #pragma unroll
  for (int j = 0; j < 8; ++j) {
    const float nm    = fmaxf(m[j], mx[j]);
    const float alpha = exp2f(m[j] - nm);   // first iter: exp2(-big)=0
    #pragma unroll
    for (int t = 0; t < 4; ++t)
      S[t][j] = exp2f(S[t][j] - nm);        // masked cols underflow to 0
    m[j] = nm;
    #pragma unroll
    for (int ct = 0; ct < 4; ++ct) Oacc[ct][j] *= alpha;
    Lacc[j] *= alpha;                       // denominator rescaled identically
  }

  // ---- P (C layout) -> LDS row-major -> reload as A-fragments ----
  #pragma unroll
  for (int t = 0; t < 4; ++t)
    #pragma unroll
    for (int j = 0; j < 8; ++j)
      pw[(j + 8*(lane >> 4))*KB + t*16 + (lane & 15)] = (_Float16)S[t][j];

  asm volatile("s_wait_dscnt 0" ::: "memory");  // order cross-lane LDS RAW in-wave

  #pragma unroll
  for (int c = 0; c < 2; ++c) {
    const _Float16* p = &pw[(lane & 15)*KB + c*32 + hi8];
    v8h lo = *(const v8h*)p;
    v8h hi = *(const v8h*)(p + 16);
    const v16h aP = __builtin_shufflevector(lo, hi, 0,1,2,3,4,5,6,7,8,9,10,11,12,13,14,15);
    // denominator: Lacc += P x ones  (every column = row sum, f32 accumulate)
    Lacc = __builtin_amdgcn_wmma_f32_16x16x32_f16(false, aP, false, bones,
                                                  (short)0, Lacc, false, false);
    #pragma unroll
    for (int ct = 0; ct < 4; ++ct) {
      // B-frag: lane holds d = ct*16 + lane%16; elements key in table order
      v16h bv;
      #pragma unroll
      for (int s = 0; s < 16; ++s)
        bv[s] = sV[(c*32 + hi8 + (s&7) + ((s>>3)<<4))*D_ + ct*16 + (lane & 15)];
      Oacc[ct] = __builtin_amdgcn_wmma_f32_16x16x32_f16(false, aP, false, bv,
                                                        (short)0, Oacc[ct], false, false);
    }
  }
}

__global__ __launch_bounds__(256)
void scan_attn_wmma(const float* __restrict__ qs, const float* __restrict__ ks,
                    const float* __restrict__ vs, const int* __restrict__ vlen,
                    float* __restrict__ out)
{
  // 48 KB total: Q region (16 KB, reused as P scratch once aQ frags are in
  // registers) + double-buffered K^T (2x8 KB) + double-buffered V (2x8 KB).
  __shared__ __align__(32) _Float16 smem[QT*D_ + 2*D_*KB + 2*KB*D_];
  _Float16* const sQ   = smem;                 // [128][64], dead after aQ load
  _Float16* const sKTb = smem + QT*D_;         // [2][64][KB]
  _Float16* const sVb  = sKTb + 2*D_*KB;       // [2][KB][64]
  _Float16* const sP   = smem;                 // aliases sQ (barrier-separated)

  const int tid  = threadIdx.x;
  const int lane = tid & 31;
  const int wave = tid >> 5;
  const int qt = blockIdx.x, h = blockIdx.y, b = blockIdx.z;
  const int q0 = qt * QT;
  const int valid = vlen[b];

  // ---- stage Q tile once: fp32 -> (scale*log2e) f16, row-major ----
  {
    const int row = tid >> 1;
    const int d0  = (tid & 1) * 32;
    const float4* src = (const float4*)(qs + (((long)(b*Q_ + q0 + row))*H_ + h)*D_ + d0);
    v8h* dst = (v8h*)&sQ[row * D_ + d0];
    #pragma unroll
    for (int i = 0; i < 4; ++i) {
      float4 f0 = src[2*i+0];
      float4 f1 = src[2*i+1];
      v8h hv;
      hv[0]=(_Float16)(f0.x*QSCALE); hv[1]=(_Float16)(f0.y*QSCALE);
      hv[2]=(_Float16)(f0.z*QSCALE); hv[3]=(_Float16)(f0.w*QSCALE);
      hv[4]=(_Float16)(f1.x*QSCALE); hv[5]=(_Float16)(f1.y*QSCALE);
      hv[6]=(_Float16)(f1.z*QSCALE); hv[7]=(_Float16)(f1.w*QSCALE);
      dst[i] = hv;
    }
  }
  __syncthreads();

  // ---- resident Q A-fragments (ISA 16-bit A layout); sQ is dead afterwards ----
  const int hi8 = (lane >> 4) * 8;
  v16h aQ[2];
  {
    const int r = wave*16 + (lane & 15);
    #pragma unroll
    for (int c = 0; c < 2; ++c) {
      const _Float16* p = &sQ[r * D_ + c*32 + hi8];
      v8h lo = *(const v8h*)p;
      v8h hi = *(const v8h*)(p + 16);
      aQ[c] = __builtin_shufflevector(lo, hi, 0,1,2,3,4,5,6,7,8,9,10,11,12,13,14,15);
    }
  }

  // all-ones B matrix for the denominator GEMM
  v16h bones;
  #pragma unroll
  for (int i = 0; i < 16; ++i) bones[i] = (_Float16)1.0f;

  v8f Oacc[4] = {};
  v8f Lacc = {};
  float m[8];
  #pragma unroll
  for (int j = 0; j < 8; ++j) m[j] = NEG_BIG;

  const int  nblk = (valid + KB - 1) / KB;    // skip fully-masked KV blocks
  _Float16* const pw = &sP[wave * 16 * KB];
  const long kvbase = (long)b * K_;

  stage_kv(ks, vs, sKTb, sVb, tid, kvbase, h);   // block 0 -> buffer 0
  __syncthreads();

  for (int blk = 0; blk < nblk; ++blk) {
    const int kb  = blk * KB;
    const int cur = blk & 1;

    // stage next block into the other buffer while computing this one
    if (blk + 1 < nblk)
      stage_kv(ks, vs, sKTb + (cur^1)*(D_*KB), sVb + (cur^1)*(KB*D_),
               tid, kvbase + kb + KB, h);
    // prefetch block n+2 into the cache hierarchy (global_prefetch_b8)
    if (blk + 2 < nblk) {
      const float* nk = ks + ((kvbase + kb + 2*KB + (tid >> 2))*H_ + h)*D_;
      __builtin_prefetch(nk, 0, 1);
      const float* nv = vs + ((kvbase + kb + 2*KB + (tid >> 2))*H_ + h)*D_;
      __builtin_prefetch(nv, 0, 1);
    }

    const _Float16* sKT = sKTb + cur*(D_*KB);
    const _Float16* sV  = sVb  + cur*(KB*D_);
    if (kb + KB <= valid)
      kv_block<false>(sKT, sV, pw, aQ, bones, Oacc, Lacc, m, lane, hi8, kb, valid);
    else
      kv_block<true >(sKT, sV, pw, aQ, bones, Oacc, Lacc, m, lane, hi8, kb, valid);

    __syncthreads();  // staging of blk+1 done; everyone finished reading cur
  }

  // ---- normalize by softmax denominator and store fp32 output ----
  #pragma unroll
  for (int j = 0; j < 8; ++j) {
    const float inv = 1.0f / Lacc[j];          // every row has >=1 valid key
    const int row = q0 + wave*16 + j + 8*(lane >> 4);
    float* orow = out + (((long)(b*Q_ + row))*H_ + h)*D_ + (lane & 15);
    #pragma unroll
    for (int ct = 0; ct < 4; ++ct)
      orow[ct*16] = Oacc[ct][j] * inv;
  }
}

extern "C" void kernel_launch(void* const* d_in, const int* in_sizes, int n_in,
                              void* d_out, int out_size, void* d_ws, size_t ws_size,
                              hipStream_t stream)
{
  (void)in_sizes; (void)n_in; (void)d_ws; (void)ws_size; (void)out_size;
  const float* qs = (const float*)d_in[0];
  const float* ks = (const float*)d_in[1];
  const float* vs = (const float*)d_in[2];
  const int*   vl = (const int*)d_in[3];
  float* out = (float*)d_out;

  dim3 grid(Q_ / QT, H_, B_);   // (16, 16, 2) = 512 workgroups
  dim3 block(256);              // 8 wave32 waves
  scan_attn_wmma<<<grid, block, 0, stream>>>(qs, ks, vs, vl, out);
}
